// SoftmaxAttentionLayer_84885733638833
// MI455X (gfx1250) — matrix operand order repaired and checked by reading, not compiled, not run on hardware
//
#include <hip/hip_runtime.h>
#include <hip/hip_bf16.h>

// ---------------------------------------------------------------------------
// Causal attention layer for MI455X (gfx1250, wave32, WMMA 16x16x32 f16).
//   B=2, S=2048, H=1024, NUM_HEADS=16, HD=64.
// ~65 GFLOP vs ~50 MB of traffic (fits in 192 MB L2) -> compute-bound on the
// matrix pipes. All matmuls via v_wmma_f32_16x16x32_f16 w/ f32 accumulation.
// GEMM staging uses the Tensor Data Mover (tensor_load_to_lds + TENSORcnt)
// with double-buffered LDS, so the inner loop is pure ds_load_b128 + wmma.
// ---------------------------------------------------------------------------

#define BDIM   2
#define SDIM   2048
#define HDIM   1024
#define H3DIM  3072
#define NHEADS 16
#define HD     64
#define BS     (BDIM * SDIM)      // 4096 rows
#define SCALE  0.125f             // 1/sqrt(64)

typedef __attribute__((ext_vector_type(16))) _Float16 v16h;
typedef __attribute__((ext_vector_type(8)))  _Float16 v8h;
typedef __attribute__((ext_vector_type(4)))  _Float16 v4h;
typedef __attribute__((ext_vector_type(4)))  float    v4f;
typedef __attribute__((ext_vector_type(8)))  float    v8f;
typedef __attribute__((ext_vector_type(4)))  unsigned int u32x4;
typedef __attribute__((ext_vector_type(8)))  int      i32x8;
typedef __attribute__((ext_vector_type(4)))  int      i32x4;

#define NEG_INF (-__builtin_inff())

// Build a 16-half fragment from two contiguous 16-byte chunks.
__device__ __forceinline__ v16h ld_frag(const _Float16* p0, const _Float16* p1) {
    v8h lo = *(const v8h*)p0;
    v8h hi = *(const v8h*)p1;
    return __builtin_shufflevector(lo, hi, 0, 1, 2, 3, 4, 5, 6, 7,
                                           8, 9, 10, 11, 12, 13, 14, 15);
}

__device__ __forceinline__ float redMax16(float v) {
    #pragma unroll
    for (int m = 1; m < 16; m <<= 1) v = fmaxf(v, __shfl_xor(v, m, 32));
    return v;
}
__device__ __forceinline__ float redSum16(float v) {
    #pragma unroll
    for (int m = 1; m < 16; m <<= 1) v += __shfl_xor(v, m, 32);
    return v;
}

// ---------------------------------------------------------------------------
// TDM: DMA a 2-D f16 tile (rows x 32 elems, row stride = strideElems) from
// global memory into LDS at byte offset lds_off. D# bitfields per CDNA5 ISA
// CDNA5 ISA (group0: count/lds/global/type, group1: sizes/strides).
// Wave-level op, tracked by TENSORcnt. This toolchain exposes the 6-arg
// builtin: (u32x4 g0, i32x8 g1, i32x4 g2, i32x4 g3, i32x8, i32 cpol).
// ---------------------------------------------------------------------------
__device__ __forceinline__ void tdm_load_tile_f16(unsigned lds_off,
                                                  const _Float16* gptr,
                                                  unsigned rows,
                                                  unsigned strideElems,
                                                  unsigned tensorRows) {
    const unsigned long long ga = (unsigned long long)(const void*)gptr;
    u32x4 g0;
    g0[0] = 1u;                                     // count=1, user descriptor
    g0[1] = lds_off;                                // lds_addr (bytes)
    g0[2] = (unsigned)(ga & 0xFFFFFFFFu);           // global_addr[31:0]
    g0[3] = (unsigned)((ga >> 32) & 0x01FFFFFFu)    // global_addr[56:32]
            | (2u << 30);                           // type=2 ("image")
    const unsigned td0 = 1024u;                     // tensor_dim0 (elems, >= tile reach)
    const unsigned td1 = tensorRows;                // tensor_dim1 (rows)
    i32x8 g1;
    g1[0] = (int)(1u << 16);                        // workgroup_mask=0, data_size=2B
    g1[1] = (int)((td0 & 0xFFFFu) << 16);           // tensor_dim0[15:0]
    g1[2] = (int)(((td0 >> 16) & 0xFFFFu) | ((td1 & 0xFFFFu) << 16));
    g1[3] = (int)(((td1 >> 16) & 0xFFFFu) | (32u << 16));   // tile_dim0 = 32
    g1[4] = (int)(rows & 0xFFFFu);                  // tile_dim1 = rows, tile_dim2=0
    g1[5] = (int)strideElems;                       // tensor_dim0_stride[31:0]
    g1[6] = 0;                                      // stride[47:32]=0, dim1_stride lo=0
    g1[7] = 0;
    const i32x4 gz4 = (i32x4)0;                     // groups 2/3 unused (2-D tile)
    const i32x8 gz8 = (i32x8)0;
    __builtin_amdgcn_tensor_load_to_lds(g0, g1, gz4, gz4, gz8, 0);
}

// ---------------------------------------------------------------------------
// f32 -> f16 bulk convert (n multiple of 4).
// ---------------------------------------------------------------------------
__global__ void __launch_bounds__(256)
cvt_f32_to_f16(const float* __restrict__ in, _Float16* __restrict__ out, int n) {
    const int i = blockIdx.x * blockDim.x + threadIdx.x;
    if (i * 4 >= n) return;
    const v4f v = ((const v4f*)in)[i];
    v4h o;
    #pragma unroll
    for (int j = 0; j < 4; ++j) o[j] = (_Float16)v[j];
    ((v4h*)out)[i] = o;
}

// ---------------------------------------------------------------------------
// C[M x N] = A[M x K] * W[N x K]^T   (NT GEMM, f16 in, f32 out).
// 128x128 block tile, TDM-staged through double-buffered LDS. 8 waves, each
// computing a 32x64 register tile = 8 WMMAs per 32-deep K step. All fragment
// reads are ds_load_b128 (WMMA A/B per-lane layouts are contiguous 16B runs).
// ---------------------------------------------------------------------------
__global__ void __launch_bounds__(256)
gemm_nt_f16_wmma(const _Float16* __restrict__ A, const _Float16* __restrict__ W,
                 float* __restrict__ C, int N, int K) {
    __shared__ __align__(64) _Float16 As[2][128][32];   // 2 x 8 KB
    __shared__ __align__(64) _Float16 Bs[2][128][32];   // 2 x 8 KB

    const int tid    = threadIdx.x;
    const int lane   = tid & 31;
    const int wave   = tid >> 5;
    const int laneHi = lane >> 4;          // 0: lanes 0-15, 1: lanes 16-31
    const int lane16 = lane & 15;

    const int mBase = blockIdx.y * 128;
    const int nBase = blockIdx.x * 128;

    const int m0 = (wave >> 1) * 32;       // wave's M offset inside block tile
    const int n0 = (wave & 1) * 64;        // wave's N offset inside block tile

    const unsigned ldsA = (unsigned)(unsigned long long)&As[0][0][0];
    const unsigned ldsB = (unsigned)(unsigned long long)&Bs[0][0][0];

    v8f acc[2][4];
    #pragma unroll
    for (int mt = 0; mt < 2; ++mt)
        #pragma unroll
        for (int nt = 0; nt < 4; ++nt) acc[mt][nt] = (v8f){};

    // ---- prime buffer 0 (wave 0 drives the TDM)
    if (wave == 0) {
        tdm_load_tile_f16(ldsA, A + (size_t)mBase * K, 128, (unsigned)K, 128);
        tdm_load_tile_f16(ldsB, W + (size_t)nBase * K, 128, (unsigned)K, 128);
    }

    for (int k0 = 0; k0 < K; k0 += 32) {
        const int buf = (k0 >> 5) & 1;
        if (wave == 0) __builtin_amdgcn_s_wait_tensorcnt(0);
        __syncthreads();                   // buf is ready; prior reads of buf^1 done

        if (wave == 0 && k0 + 32 < K) {    // DMA next K tile into the other buffer
            tdm_load_tile_f16(ldsA + (buf ^ 1) * 8192u,
                              A + (size_t)mBase * K + k0 + 32, 128, (unsigned)K, 128);
            tdm_load_tile_f16(ldsB + (buf ^ 1) * 8192u,
                              W + (size_t)nBase * K + k0 + 32, 128, (unsigned)K, 128);
        }

        // ---- fragments from LDS (all ds_load_b128)
        v16h aF[2];
        #pragma unroll
        for (int mt = 0; mt < 2; ++mt) {
            const _Float16* r = &As[buf][m0 + mt * 16 + lane16][0];
            aF[mt] = ld_frag(r + laneHi * 8, r + 16 + laneHi * 8);
        }
        #pragma unroll
        for (int nt = 0; nt < 4; ++nt) {
            const _Float16* r = &Bs[buf][n0 + nt * 16 + lane16][0];
            const v16h bF = ld_frag(r + laneHi * 8, r + 16 + laneHi * 8);
            #pragma unroll
            for (int mt = 0; mt < 2; ++mt)
                acc[mt][nt] = __builtin_amdgcn_wmma_f32_16x16x32_f16(
                    false, aF[mt], false, bF, (short)0, acc[mt][nt], false, false);
        }
    }

    #pragma unroll
    for (int mt = 0; mt < 2; ++mt)
        #pragma unroll
        for (int nt = 0; nt < 4; ++nt)
            #pragma unroll
            for (int i = 0; i < 8; ++i) {
                const int m = mBase + m0 + mt * 16 + (laneHi << 3) + i;
                const int n = nBase + n0 + nt * 16 + lane16;
                C[(size_t)m * N + n] = acc[mt][nt][i];
            }
}

// ---------------------------------------------------------------------------
// RoPE on q,k + split qkv (f32, [b*s][3*H]) into head-major f16 tensors
// q/k/v : [b][h][s][d].
// ---------------------------------------------------------------------------
__global__ void __launch_bounds__(256)
rope_split(const float* __restrict__ qkv, _Float16* __restrict__ qh,
           _Float16* __restrict__ kh, _Float16* __restrict__ vh) {
    const int idx = blockIdx.x * blockDim.x + threadIdx.x;   // over BS*HDIM
    if (idx >= BS * HDIM) return;
    const int d  = idx & (HD - 1);
    const int h  = (idx >> 6) & (NHEADS - 1);
    const int ns = idx >> 10;            // b*S + s
    const int s  = ns & (SDIM - 1);
    const int b  = ns >> 11;

    const float* base = qkv + (size_t)ns * H3DIM;
    const float qv = base[h * HD + d];
    const float kv = base[HDIM + h * HD + d];
    const float vv = base[2 * HDIM + h * HD + d];

    const int   dh  = d & 31;
    // inv_freq = 10000^(-2*dh/64);  log(10000) = 9.210340371976184
    const float inv = __expf(-(float)(2 * dh) * (9.210340371976184f / 64.0f));
    const float ang = (float)s * inv;
    const float cs = __cosf(ang), sn = __sinf(ang);

    const float qrot = (d < 32) ? -base[h * HD + d + 32]
                                :  base[h * HD + d - 32];
    const float krot = (d < 32) ? -base[HDIM + h * HD + d + 32]
                                :  base[HDIM + h * HD + d - 32];

    const size_t o = ((size_t)(b * NHEADS + h) * SDIM + s) * HD + d;
    qh[o] = (_Float16)(qv * cs + qrot * sn);
    kh[o] = (_Float16)(kv * cs + krot * sn);
    vh[o] = (_Float16)vv;
}

// ---------------------------------------------------------------------------
// Causal flash attention. One wave owns a 16-row q tile of one (b,h).
// 32 key columns / iteration: 2x(2 chained) score WMMAs, online softmax
// (shfl reductions within 16-lane halves, matching the C-layout M split),
// f16 LDS bounce to re-layout P into the A-fragment pattern, 4 PV WMMAs.
// ---------------------------------------------------------------------------
__global__ void __launch_bounds__(256)
attn_wmma(const _Float16* __restrict__ qh, const _Float16* __restrict__ kh,
          const _Float16* __restrict__ vh, _Float16* __restrict__ aout) {
    __shared__ __align__(64) _Float16 pLds[8][16][32];

    const int lane   = threadIdx.x & 31;
    const int wave   = threadIdx.x >> 5;
    const int laneHi = lane >> 4;
    const int lane16 = lane & 15;

    const int h = blockIdx.y;
    const int b = blockIdx.z;
    const int qBase = (blockIdx.x * 8 + wave) * 16;

    const size_t head = (size_t)(b * NHEADS + h) * SDIM * HD;
    const _Float16* Q = qh + head;
    const _Float16* K = kh + head;
    const _Float16* V = vh + head;

    // Preload the 16x64 Q tile as two A-fragments (K = 0..31 and 32..63).
    const _Float16* qrow = Q + (size_t)(qBase + lane16) * HD;
    const v16h aQ0 = ld_frag(qrow + laneHi * 8,      qrow + 16 + laneHi * 8);
    const v16h aQ1 = ld_frag(qrow + 32 + laneHi * 8, qrow + 48 + laneHi * 8);

    float rowMax[8], rowSum[8];
    v8f acc[4];
    #pragma unroll
    for (int i = 0; i < 8; ++i) { rowMax[i] = NEG_INF; rowSum[i] = 0.f; }
    #pragma unroll
    for (int c = 0; c < 4; ++c) acc[c] = (v8f){};

    for (int kBase = 0; kBase <= qBase + 15; kBase += 32) {
        // ---- scores: two 16x16 tiles covering key columns kBase..kBase+31
        v8f sc[2] = {(v8f){}, (v8f){}};
        #pragma unroll
        for (int t = 0; t < 2; ++t) {
            const _Float16* krow =
                K + (size_t)(kBase + t * 16 + lane16) * HD + laneHi * 8;
            const v16h b0 = ld_frag(krow,      krow + 16);
            const v16h b1 = ld_frag(krow + 32, krow + 48);
            sc[t] = __builtin_amdgcn_wmma_f32_16x16x32_f16(false, aQ0, false, b0,
                                                           (short)0, sc[t], false, false);
            sc[t] = __builtin_amdgcn_wmma_f32_16x16x32_f16(false, aQ1, false, b1,
                                                           (short)0, sc[t], false, false);
        }

        // ---- online softmax (accumulator row i -> global row qBase+8*hi+i)
        float corr[8];
        #pragma unroll
        for (int i = 0; i < 8; ++i) {
            const int row = qBase + (laneHi << 3) + i;
            const int c0  = kBase + lane16;
            float v0 = sc[0][i] * SCALE;  if (c0      > row) v0 = NEG_INF;
            float v1 = sc[1][i] * SCALE;  if (c0 + 16 > row) v1 = NEG_INF;

            const float tmax = redMax16(fmaxf(v0, v1));
            const float nm   = fmaxf(rowMax[i], tmax);
            const float cr   = (nm == NEG_INF) ? 1.f : __expf(rowMax[i] - nm);
            const float p0   = (v0 == NEG_INF) ? 0.f : __expf(v0 - nm);
            const float p1   = (v1 == NEG_INF) ? 0.f : __expf(v1 - nm);
            rowSum[i] = rowSum[i] * cr + redSum16(p0 + p1);
            rowMax[i] = nm;
            corr[i]   = cr;

            const int m = (laneHi << 3) + i;
            pLds[wave][m][lane16]      = (_Float16)p0;
            pLds[wave][m][lane16 + 16] = (_Float16)p1;
        }
        #pragma unroll
        for (int c = 0; c < 4; ++c)
            #pragma unroll
            for (int i = 0; i < 8; ++i) acc[c][i] *= corr[i];

        // ---- P (16x32) from LDS as an A-fragment (two ds_load_b128);
        //      LDS is in-order within a wave, no block barrier needed.
        const _Float16* prow = &pLds[wave][lane16][0];
        const v16h aP = ld_frag(prow + laneHi * 16, prow + laneHi * 16 + 8);

        // ---- acc += P @ V[kBase..kBase+31][:], 4 column chunks of 16
        #pragma unroll
        for (int c = 0; c < 4; ++c) {
            v16h bV;
            #pragma unroll
            for (int j = 0; j < 16; ++j)
                bV[j] = V[(size_t)(kBase + laneHi * 16 + j) * HD + c * 16 + lane16];
            acc[c] = __builtin_amdgcn_wmma_f32_16x16x32_f16(false, aP, false, bV,
                                                            (short)0, acc[c], false, false);
        }
    }

    // ---- normalize, store f16 rows [b*s][H] for the output projection
    #pragma unroll
    for (int c = 0; c < 4; ++c)
        #pragma unroll
        for (int i = 0; i < 8; ++i) {
            const int row = qBase + (laneHi << 3) + i;
            const int col = c * 16 + lane16;
            aout[((size_t)(b * SDIM + row)) * HDIM + h * HD + col] =
                (_Float16)(acc[c][i] / rowSum[i]);
        }
}

// ---------------------------------------------------------------------------
extern "C" void kernel_launch(void* const* d_in, const int* in_sizes, int n_in,
                              void* d_out, int out_size, void* d_ws, size_t ws_size,
                              hipStream_t stream) {
    (void)in_sizes; (void)n_in; (void)out_size; (void)ws_size;
    const float* x     = (const float*)d_in[0];
    const float* w_qkv = (const float*)d_in[1];
    const float* w_o   = (const float*)d_in[2];
    float*       out   = (float*)d_out;

    char* ws = (char*)d_ws;
    size_t off = 0;
    float*    qkv   = (float*)ws;                          // BS x 3H f32 (50.3 MB)
    _Float16* aoutH = (_Float16*)ws;                       // alias (qkv dead by then)
    off += (size_t)BS * H3DIM * sizeof(float);
    _Float16* xh    = (_Float16*)(ws + off); off += (size_t)BS * HDIM * 2;
    _Float16* wqkvh = (_Float16*)(ws + off); off += (size_t)H3DIM * HDIM * 2;
    _Float16* woh   = (_Float16*)(ws + off); off += (size_t)HDIM * HDIM * 2;
    _Float16* qh    = (_Float16*)(ws + off); off += (size_t)BS * HDIM * 2;
    _Float16* kh    = (_Float16*)(ws + off); off += (size_t)BS * HDIM * 2;
    _Float16* vh    = (_Float16*)(ws + off);

    dim3 blk(256);

    // 0) one-time f16 conversions
    cvt_f32_to_f16<<<dim3((BS * HDIM) / 4 / 256), blk, 0, stream>>>(x, xh, BS * HDIM);
    cvt_f32_to_f16<<<dim3((H3DIM * HDIM) / 4 / 256), blk, 0, stream>>>(w_qkv, wqkvh, H3DIM * HDIM);
    cvt_f32_to_f16<<<dim3((HDIM * HDIM) / 4 / 256), blk, 0, stream>>>(w_o, woh, HDIM * HDIM);

    // 1) qkv = x @ w_qkv^T
    gemm_nt_f16_wmma<<<dim3(H3DIM / 128, BS / 128), blk, 0, stream>>>(
        xh, wqkvh, qkv, H3DIM, HDIM);

    // 2) RoPE + split to f16 head-major q/k/v
    rope_split<<<dim3((BS * HDIM) / 256), blk, 0, stream>>>(qkv, qh, kh, vh);

    // 3) causal flash attention (qkv buffer becomes the f16 attention output)
    attn_wmma<<<dim3(SDIM / 16 / 8, NHEADS, BDIM), blk, 0, stream>>>(qh, kh, vh, aoutH);

    // 4) out = aout @ w_o^T
    gemm_nt_f16_wmma<<<dim3(HDIM / 128, BS / 128), blk, 0, stream>>>(
        aoutH, woh, out, HDIM, HDIM);
}
